// SparseLayer_23725399343675
// MI455X (gfx1250) — matrix operand order repaired and checked by reading, not compiled, not run on hardware
//
#include <hip/hip_runtime.h>
#include <stdint.h>

// ---- CDNA5 types -----------------------------------------------------------
typedef __attribute__((ext_vector_type(16))) __bf16       v16bf;
typedef __attribute__((ext_vector_type(8)))  float        v8f;
typedef __attribute__((ext_vector_type(4)))  unsigned int v4u;
typedef __attribute__((ext_vector_type(8)))  int          v8i;
typedef __attribute__((ext_vector_type(4)))  int          v4i;

#define SIZE_IN  4096   // K
#define SIZE_OUT 4096   // M
#define BATCH    16384  // N
#define BK       32     // K step == WMMA K (bf16)
#define BKP      40     // padded LDS row in shorts (80B, 16B-aligned)

union FragU { v16bf v; v4u u[2]; };

__device__ __forceinline__ uint32_t pack2_bf16(float lo, float hi) {
  uint32_t a = __float_as_uint(lo);
  uint32_t b = __float_as_uint(hi);
  a += 0x7fffu + ((a >> 16) & 1u);
  b += 0x7fffu + ((b >> 16) & 1u);
  return (a >> 16) | (b & 0xffff0000u);
}
__device__ __forceinline__ unsigned short f32_bf16(float f) {
  uint32_t u = __float_as_uint(f);
  u += 0x7fffu + ((u >> 16) & 1u);
  return (unsigned short)(u >> 16);
}

// ======================= Phase 1: transpose + fp32->bf16 ====================
// in: [R, C] fp32 (C contiguous)  ->  out: [C, R] bf16 (R contiguous)
__global__ __launch_bounds__(256)
void transpose_cvt_bf16(const float* __restrict__ in,
                        unsigned short* __restrict__ out, int R, int C) {
  __shared__ float tile[32][33];
  const int c0 = blockIdx.x * 32;
  const int r0 = blockIdx.y * 32;
  const int tc = threadIdx.x & 31;
  const int t8 = threadIdx.x >> 5;     // 0..7
  #pragma unroll
  for (int q = 0; q < 4; ++q) {
    int rr = t8 + q * 8;
    tile[rr][tc] = in[(size_t)(r0 + rr) * C + (c0 + tc)];
  }
  __syncthreads();
  #pragma unroll
  for (int q = 0; q < 4; ++q) {
    int cc = t8 + q * 8;
    out[(size_t)(c0 + cc) * R + (r0 + tc)] = f32_bf16(tile[tc][cc]);
  }
}

// ======================= Phase 2: TDM-fed bf16 WMMA GEMM ====================
#define TBM 128
#define TBN 256
#define T_TILES_M (SIZE_OUT / TBM)   // 32
#define T_TILES_N (BATCH / TBN)      // 64
#define T_GROUP   8                  // N-tiles per L2 pass (Wt 32MB + x panel in L2)

// Issue one TDM 2D tile load: tile_rows x BK bf16, K contiguous in global,
// landing in LDS with rows padded 64B+16B (pad_interval=16dw, pad_amount=4dw)
// to match the BKP=40-short LDS rows the fragment ds_load_b128s expect.
__device__ __forceinline__ void tdm_load_tile(uint32_t lds_addr,
                                              const unsigned short* gptr,
                                              int tile_rows) {
  uint64_t ga = (uint64_t)(uintptr_t)gptr;
  v4u g0;
  g0[0] = 1u;                                               // count=1, user D#
  g0[1] = lds_addr;                                         // lds_addr (bytes)
  g0[2] = (uint32_t)ga;                                     // global_addr[31:0]
  g0[3] = ((uint32_t)(ga >> 32) & 0x01FFFFFFu) | (2u << 30);// addr[56:32]|type=2
  v8i g1;
  g1[0] = (int)((1u << 16)      // data_size = 2B
              | (1u << 20)      // pad_enable
              | (3u << 22)      // pad_interval: 16 dwords (one 64B tile row)
              | (3u << 25));    // pad_amount: 4 dwords (16B) -> 80B LDS rows
  g1[1] = (int)((uint32_t)SIZE_IN << 16);   // abar=0 | tensor_dim0 lo16 = 4096
  g1[2] = (int)(0x8000u << 16);             // td0 hi16=0 | tensor_dim1 lo16
  g1[3] = (int)((uint32_t)BK << 16);        // td1 hi16=0 | tile_dim0 = 32
  g1[4] = tile_rows;                        // tile_dim1 | tile_dim2 = 0
  g1[5] = SIZE_IN;                          // tensor_dim0_stride lo32 = 4096
  g1[6] = 0;                                // stride0 hi | stride1 lo
  g1[7] = 0;
  v4i gz4 = {0, 0, 0, 0};                   // 2D tile: groups 2/3 unused
  v8i gz8 = {0, 0, 0, 0, 0, 0, 0, 0};       // 6-arg form: zero-filled extra group
  __builtin_amdgcn_tensor_load_to_lds(g0, g1, gz4, gz4, gz8, 0);
}

__global__ __launch_bounds__(256)
void gemm_tdm_bf16_wmma(const unsigned short* __restrict__ Wt,  // [M,K] bf16
                        const unsigned short* __restrict__ Xt,  // [N,K] bf16
                        float* __restrict__ Out) {              // [M,N] f32
  __shared__ unsigned short As[2][TBM][BKP];
  __shared__ unsigned short Bs[2][TBN][BKP];

  const int tid  = threadIdx.x;
  const int lane = tid & 31;
  const int wave = tid >> 5;
  const int wm   = wave >> 2;   // 0..1 -> 64 rows each
  const int wn   = wave & 3;    // 0..3 -> 64 cols each

  const int per_grp = T_TILES_M * T_GROUP;
  const int g  = blockIdx.x / per_grp;
  const int r  = blockIdx.x % per_grp;
  const int tn = g * T_GROUP + (r / T_TILES_M);
  const int tm = r % T_TILES_M;
  const int m0 = tm * TBM;
  const int n0 = tn * TBN;

  const unsigned short* gA = Wt + (size_t)m0 * SIZE_IN;
  const unsigned short* gB = Xt + (size_t)n0 * SIZE_IN;

  v8f acc[4][4];
  #pragma unroll
  for (int i = 0; i < 4; ++i)
    #pragma unroll
    for (int j = 0; j < 4; ++j)
      #pragma unroll
      for (int e = 0; e < 8; ++e) acc[i][j][e] = 0.0f;

  // ISA 16-bit fragment addressing (wave32):
  //  A 16x32: lanes 0-15 K{0-7,16-23}, lanes 16-31 K{8-15,24-31}
  //  B 32x16: lanes 0-15 K{0-7,16-23}, lanes 16-31 K{8-15,24-31}
  const int half = lane >> 4;
  const int l16  = lane & 15;
  const int kbA  = half * 8;
  const int kbB  = half * 8;

  if (wave == 0) {   // prologue: stage K-tile 0 (TENSORcnt = 2)
    tdm_load_tile((uint32_t)(uintptr_t)&As[0][0][0], gA, TBM);
    tdm_load_tile((uint32_t)(uintptr_t)&Bs[0][0][0], gB, TBN);
  }

  const int NT = SIZE_IN / BK;   // 128
  for (int kt = 0; kt < NT; ++kt) {
    const int cur = kt & 1;
    __syncthreads();   // everyone done reading buf[cur^1] before its refill
    if (wave == 0) {
      if (kt + 1 < NT) {
        tdm_load_tile((uint32_t)(uintptr_t)&As[cur ^ 1][0][0],
                      gA + (size_t)(kt + 1) * BK, TBM);
        tdm_load_tile((uint32_t)(uintptr_t)&Bs[cur ^ 1][0][0],
                      gB + (size_t)(kt + 1) * BK, TBN);
        __builtin_amdgcn_s_wait_tensorcnt(2);   // current pair landed
      } else {
        __builtin_amdgcn_s_wait_tensorcnt(0);
      }
    }
    __syncthreads();   // buf[cur] visible to all waves

    FragU a[4], b[4];
    #pragma unroll
    for (int i = 0; i < 4; ++i) {
      int am = wm * 64 + i * 16 + l16;
      a[i].u[0] = *(const v4u*)&As[cur][am][kbA];
      a[i].u[1] = *(const v4u*)&As[cur][am][kbA + 16];
    }
    #pragma unroll
    for (int j = 0; j < 4; ++j) {
      int bn = wn * 64 + j * 16 + l16;
      b[j].u[0] = *(const v4u*)&Bs[cur][bn][kbB];
      b[j].u[1] = *(const v4u*)&Bs[cur][bn][kbB + 16];
    }
    #pragma unroll
    for (int i = 0; i < 4; ++i)
      #pragma unroll
      for (int j = 0; j < 4; ++j)
        acc[i][j] = __builtin_amdgcn_wmma_f32_16x16x32_bf16(
            false, a[i].v, false, b[j].v, (short)0, acc[i][j], false, false);
  }

  const int mrow = half * 8;
  #pragma unroll
  for (int i = 0; i < 4; ++i) {
    #pragma unroll
    for (int j = 0; j < 4; ++j) {
      int ng = n0 + wn * 64 + j * 16 + l16;
      int mb = m0 + wm * 64 + i * 16 + mrow;
      #pragma unroll
      for (int e = 0; e < 8; ++e)
        Out[(size_t)(mb + e) * BATCH + ng] = acc[i][j][e];
    }
  }
}

// ================= Fallback: fused convert-in-kernel GEMM (round 1) =========
__global__ __launch_bounds__(256)
void wt_x_gemm_bf16_fused(const float* __restrict__ X,
                          const float* __restrict__ W,
                          float* __restrict__ Out) {
  enum { FBM = 128, FBN = 128 };
  __shared__ unsigned short As[FBM][BKP];
  __shared__ unsigned short Bs[FBN][BKP];

  const int tid  = threadIdx.x;
  const int lane = tid & 31;
  const int wave = tid >> 5;
  const int wm   = wave >> 2;
  const int wn   = wave & 3;

  const int tiles_m = SIZE_OUT / FBM;     // 32
  const int per_grp = tiles_m * 8;
  const int g  = blockIdx.x / per_grp;
  const int r  = blockIdx.x % per_grp;
  const int tn = g * 8 + (r / tiles_m);
  const int tm = r % tiles_m;
  const int m0 = tm * FBM;
  const int n0 = tn * FBN;

  v8f acc[4][2];
  #pragma unroll
  for (int i = 0; i < 4; ++i)
    #pragma unroll
    for (int j = 0; j < 2; ++j)
      #pragma unroll
      for (int e = 0; e < 8; ++e) acc[i][j][e] = 0.0f;

  const int half = lane >> 4;
  const int l16  = lane & 15;
  const int kbA  = half * 8;
  const int kbB  = half * 8;

  for (int k0 = 0; k0 < SIZE_IN; k0 += BK) {
    __syncthreads();
    #pragma unroll
    for (int it = 0; it < (FBM * BK / 2) / 256; ++it) {
      int p = it * 256 + tid;
      int m = p & (FBM - 1);
      int k = (p >> 7) * 2;
      float a0 = W[(size_t)(k0 + k)     * SIZE_OUT + (m0 + m)];
      float a1 = W[(size_t)(k0 + k + 1) * SIZE_OUT + (m0 + m)];
      *(uint32_t*)&As[m][k] = pack2_bf16(a0, a1);
    }
    #pragma unroll
    for (int it = 0; it < (FBN * BK / 2) / 256; ++it) {
      int p = it * 256 + tid;
      int n = p & (FBN - 1);
      int k = (p >> 7) * 2;
      float b0 = X[(size_t)(k0 + k)     * BATCH + (n0 + n)];
      float b1 = X[(size_t)(k0 + k + 1) * BATCH + (n0 + n)];
      *(uint32_t*)&Bs[n][k] = pack2_bf16(b0, b1);
    }
    if (k0 + BK < SIZE_IN) {
      __builtin_prefetch(&W[(size_t)(k0 + BK) * SIZE_OUT + m0 + tid], 0, 1);
      __builtin_prefetch(&X[(size_t)(k0 + BK) * BATCH    + n0 + tid], 0, 1);
    }
    __syncthreads();

    FragU a[4], b[2];
    #pragma unroll
    for (int i = 0; i < 4; ++i) {
      int am = wm * 64 + i * 16 + l16;
      a[i].u[0] = *(const v4u*)&As[am][kbA];
      a[i].u[1] = *(const v4u*)&As[am][kbA + 16];
    }
    #pragma unroll
    for (int j = 0; j < 2; ++j) {
      int bn = wn * 32 + j * 16 + l16;
      b[j].u[0] = *(const v4u*)&Bs[bn][kbB];
      b[j].u[1] = *(const v4u*)&Bs[bn][kbB + 16];
    }
    #pragma unroll
    for (int i = 0; i < 4; ++i)
      #pragma unroll
      for (int j = 0; j < 2; ++j)
        acc[i][j] = __builtin_amdgcn_wmma_f32_16x16x32_bf16(
            false, a[i].v, false, b[j].v, (short)0, acc[i][j], false, false);
  }

  const int mrow = half * 8;
  #pragma unroll
  for (int i = 0; i < 4; ++i) {
    #pragma unroll
    for (int j = 0; j < 2; ++j) {
      int ng = n0 + wn * 32 + j * 16 + l16;
      int mb = m0 + wm * 64 + i * 16 + mrow;
      #pragma unroll
      for (int e = 0; e < 8; ++e)
        Out[(size_t)(mb + e) * BATCH + ng] = acc[i][j][e];
    }
  }
}

// ============================ host-side launch ==============================
extern "C" void kernel_launch(void* const* d_in, const int* in_sizes, int n_in,
                              void* d_out, int out_size, void* d_ws, size_t ws_size,
                              hipStream_t stream) {
  (void)in_sizes; (void)n_in; (void)out_size;
  const float* x = (const float*)d_in[0];   // [SIZE_IN, BATCH]
  const float* W = (const float*)d_in[1];   // [SIZE_IN, SIZE_OUT]
  float* out = (float*)d_out;               // [SIZE_OUT, BATCH]

  const size_t needW = (size_t)SIZE_OUT * SIZE_IN * sizeof(unsigned short); // 32MB
  const size_t needX = (size_t)BATCH   * SIZE_IN * sizeof(unsigned short);  // 128MB

  if (ws_size >= needW + needX) {
    unsigned short* Wt = (unsigned short*)d_ws;                 // [M,K] bf16
    unsigned short* Xt = Wt + (size_t)SIZE_OUT * SIZE_IN;       // [N,K] bf16
    dim3 gw(SIZE_OUT / 32, SIZE_IN / 32);
    transpose_cvt_bf16<<<gw, 256, 0, stream>>>(W, Wt, SIZE_IN, SIZE_OUT);
    dim3 gx(BATCH / 32, SIZE_IN / 32);
    transpose_cvt_bf16<<<gx, 256, 0, stream>>>(x, Xt, SIZE_IN, BATCH);
    const int tiles = T_TILES_M * T_TILES_N;   // 2048
    gemm_tdm_bf16_wmma<<<tiles, 256, 0, stream>>>(Wt, Xt, out);
  } else {
    const int tiles = (SIZE_OUT / 128) * (BATCH / 128);         // 4096
    wt_x_gemm_bf16_fused<<<tiles, 256, 0, stream>>>(x, W, out);
  }
}
